// model_base_79379585564695
// MI455X (gfx1250) — matrix operand order repaired and checked by reading, not compiled, not run on hardware
//
#include <hip/hip_runtime.h>

// Graph-attention layer, MI455X (gfx1250, wave32, WMMA bf16 16x16x32).
// Structured softmax: each attention row has only 3 distinct values
// (non-edge -> exp(0)=1, off-diag edge -> exp(leaky(zi)), diag edge ->
// exp(leaky(zi+zj))), so the N x N softmax + attn@z collapses to ONE
// streaming pass over adj (adj@z via WMMA + exact fp32 row degrees) plus
// closed-form per-row scalar math.  HBM-bound: ~256 MB / 23.3 TB/s ~= 11 us.
// K is split 4-ways in the big GEMM for 2048 concurrent waves (latency
// hiding for the streaming loads); partials are folded in the epilogue.

#define N_ROWS 8192
#define K_IN   512
#define N_OUT  64
#define SLOPE  0.01f
#define KSPLIT 4
#define KSEG   (N_ROWS / KSPLIT)   // 2048

typedef __attribute__((ext_vector_type(16))) __bf16 v16bf;
typedef __attribute__((ext_vector_type(8)))  __bf16 v8bf;
typedef __attribute__((ext_vector_type(8)))  float  v8f;
typedef __attribute__((ext_vector_type(4)))  float  f4;

// Fragment layouts used here (wave32; contiguous k, same order in A and B):
//  A (16x32 MxK, bf16): lane L -> M=L&15, g=L>>4; elems 0..15 = K[16g..16g+15]
//  B (32x16 KxN, bf16): lane L -> N=L&15, g=L>>4; elems 0..15 = K[16g..16g+15]
//  C/D (16x16 f32):     VGPR r -> M = r + 8g, N = L&15

// ---------------------------------------------------------------------------
// Kernel 1: z = X @ W^T + b   (fp32 out, plus bf16 transposed copy zbt[c][i])
// ---------------------------------------------------------------------------
__global__ __launch_bounds__(256) void gemm1_z_kernel(
    const float* __restrict__ X, const float* __restrict__ W,
    const float* __restrict__ bias,
    float* __restrict__ z, __bf16* __restrict__ zbt)
{
  const int lane = threadIdx.x & 31;
  const int wave = threadIdx.x >> 5;
  const int m0   = (blockIdx.x * 8 + wave) * 16;
  const int m    = lane & 15;     // row within tile (A) / col within tile (B)
  const int g    = lane >> 4;

  v8f acc[4] = {};

  const float* xrow = X + (size_t)(m0 + m) * K_IN + 16 * g;

  for (int kb = 0; kb < K_IN; kb += 32) {
    v16bf a;
#pragma unroll
    for (int i = 0; i < 8; ++i) a[i]     = (__bf16)xrow[kb + i];
#pragma unroll
    for (int i = 0; i < 8; ++i) a[8 + i] = (__bf16)xrow[kb + 8 + i];

#pragma unroll
    for (int t = 0; t < 4; ++t) {
      // B[k][n] = W[n][k]; W row-major (64 x 512) => contiguous 16-float run
      const float* wrow = W + (size_t)(t * 16 + m) * K_IN + kb + 16 * g;
      v16bf bf;
#pragma unroll
      for (int i = 0; i < 16; ++i) bf[i] = (__bf16)wrow[i];
      acc[t] = __builtin_amdgcn_wmma_f32_16x16x32_bf16(
          false, a, false, bf, (short)0, acc[t], false, false);
    }
  }

#pragma unroll
  for (int t = 0; t < 4; ++t) {
    const int col = t * 16 + m;
    const float bb = bias[col];
    v8bf pk;
#pragma unroll
    for (int r = 0; r < 8; ++r) {
      const int row = m0 + g * 8 + r;
      const float v = acc[t][r] + bb;
      z[(size_t)row * N_OUT + col] = v;
      pk[r] = (__bf16)v;
    }
    // transposed bf16 store: rows m0+8g..+7 are contiguous in zbt
    *(v8bf*)(zbt + (size_t)col * N_ROWS + m0 + g * 8) = pk;
  }
}

// ---------------------------------------------------------------------------
// Kernel 2: Ep[seg] = adj[:, seg] @ z[seg]  and  dP[seg] = rowsum(adj[:, seg])
// (WMMA bf16; adj in {0,1} is exact in bf16, degrees exact in fp32 VALU.)
// adj streamed once with non-temporal loads; zbt (1 MB) stays L2-resident.
// Grid: 64 M-blocks x KSPLIT K-segments -> 2048 waves.
// ---------------------------------------------------------------------------
__global__ __launch_bounds__(256) void gemm2_attn_kernel(
    const float* __restrict__ adj, const __bf16* __restrict__ zbt,
    float* __restrict__ Ep, float* __restrict__ dsumP)
{
  const int lane = threadIdx.x & 31;
  const int wave = threadIdx.x >> 5;
  const int mblk = blockIdx.x & 63;
  const int seg  = blockIdx.x >> 6;
  const int m0   = (mblk * 8 + wave) * 16;
  const int m    = lane & 15;
  const int g    = lane >> 4;
  const int k0   = seg * KSEG;

  const float*  arow = adj + (size_t)(m0 + m) * N_ROWS + k0 + 16 * g;
  const __bf16* zb   = zbt + k0 + 16 * g;

  float rs = 0.0f;
  v8f acc[4] = {};

  for (int kb = 0; kb < KSEG; kb += 32) {
    __builtin_prefetch(arow + kb + 1024, 0, 0);   // global_prefetch ahead
    f4 x0 = __builtin_nontemporal_load((const f4*)(arow + kb));
    f4 x1 = __builtin_nontemporal_load((const f4*)(arow + kb + 4));
    f4 x2 = __builtin_nontemporal_load((const f4*)(arow + kb + 8));
    f4 x3 = __builtin_nontemporal_load((const f4*)(arow + kb + 12));
    v16bf a;
#pragma unroll
    for (int i = 0; i < 4; ++i) {
      a[i]      = (__bf16)x0[i];
      a[4 + i]  = (__bf16)x1[i];
      a[8 + i]  = (__bf16)x2[i];
      a[12 + i] = (__bf16)x3[i];
      rs += x0[i] + x1[i] + x2[i] + x3[i];
    }
#pragma unroll
    for (int t = 0; t < 4; ++t) {
      const v16bf bf = *(const v16bf*)(zb + (size_t)(t * 16 + m) * N_ROWS + kb);
      acc[t] = __builtin_amdgcn_wmma_f32_16x16x32_bf16(
          false, a, false, bf, (short)0, acc[t], false, false);
    }
  }

  // lanes m and m+16 hold disjoint K halves of the same row
  rs += __shfl_xor(rs, 16, 32);
  if (g == 0) dsumP[seg * N_ROWS + m0 + m] = rs;

#pragma unroll
  for (int t = 0; t < 4; ++t)
#pragma unroll
    for (int r = 0; r < 8; ++r)
      Ep[((size_t)seg * N_ROWS + m0 + g * 8 + r) * N_OUT + t * 16 + m] =
          acc[t][r];
}

// ---------------------------------------------------------------------------
// Kernel 3: zi = z.a1, zj = z.a2 per row
// ---------------------------------------------------------------------------
__global__ __launch_bounds__(256) void row_zizj_kernel(
    const float* __restrict__ z, const float* __restrict__ a1,
    const float* __restrict__ a2, float* __restrict__ zi, float* __restrict__ zj)
{
  const int i = blockIdx.x * blockDim.x + threadIdx.x;
  if (i >= N_ROWS) return;
  const float* zr = z + (size_t)i * N_OUT;
  float s1 = 0.0f, s2 = 0.0f;
#pragma unroll 8
  for (int c = 0; c < N_OUT; ++c) {
    s1 = fmaf(a1[c], zr[c], s1);
    s2 = fmaf(a2[c], zr[c], s2);
  }
  zi[i] = s1;
  zj[i] = s2;
}

// ---------------------------------------------------------------------------
// Kernel 4a/4b: Sall[c] = sum_i z[i][c]  (64-block partials, then tiny fold)
// ---------------------------------------------------------------------------
__global__ __launch_bounds__(256) void colsum_part_kernel(
    const float* __restrict__ z, float* __restrict__ SallP)
{
  __shared__ float red[256];
  const int t = threadIdx.x;
  const int c = t & 63;
  const int p = t >> 6;                 // 4 partials per column
  const int base = blockIdx.x * 128;    // 128 rows per block
  float s = 0.0f;
  for (int i = p; i < 128; i += 4)
    s += z[(size_t)(base + i) * N_OUT + c];
  red[t] = s;
  __syncthreads();
  if (t < 64)
    SallP[blockIdx.x * 64 + c] = red[c] + red[c + 64] + red[c + 128] + red[c + 192];
}

__global__ __launch_bounds__(64) void colsum_final_kernel(
    const float* __restrict__ SallP, float* __restrict__ Sall)
{
  const int c = threadIdx.x;
  float s = 0.0f;
#pragma unroll 8
  for (int b = 0; b < 64; ++b) s += SallP[b * 64 + c];
  Sall[c] = s;
}

// ---------------------------------------------------------------------------
// Kernel 5: fold K-split partials + closed-form softmax aggregate + ReLU
// ---------------------------------------------------------------------------
__device__ __forceinline__ float leaky(float x) {
  return x > 0.0f ? x : SLOPE * x;
}

__global__ __launch_bounds__(64) void finalize_kernel(
    const float* __restrict__ z, const float* __restrict__ Ep,
    const float* __restrict__ dsumP, const float* __restrict__ zi,
    const float* __restrict__ zj, const float* __restrict__ Sall,
    const float* __restrict__ adj, const float* __restrict__ eye,
    float* __restrict__ out)
{
  const int i = blockIdx.x;
  const int c = threadIdx.x;

  float di = 0.0f, Ei = 0.0f;
#pragma unroll
  for (int s = 0; s < KSPLIT; ++s) {
    di += dsumP[s * N_ROWS + i];
    Ei += Ep[((size_t)s * N_ROWS + i) * N_OUT + c];
  }

  const float ziv = zi[i];
  const float zjv = zj[i];
  const float aii = adj[(size_t)i * N_ROWS + i];
  const float eii = eye[(size_t)i * N_ROWS + i];

  const float el = expf(leaky(ziv));                         // off-diag edges
  const float ed = expf(leaky(aii * ziv + aii * eii * zjv)); // diagonal entry
  const float denom =
      ((float)N_ROWS - di) + (di - aii) * el + aii * ed;     // non-edges: exp(0)=1

  const float zc = z[(size_t)i * N_OUT + c];
  const float numer = (Sall[c] - Ei) + el * (Ei - aii * zc) + aii * ed * zc;
  const float h = zc - numer / denom;
  out[(size_t)i * N_OUT + c] = fmaxf(h, 0.0f);
}

// ---------------------------------------------------------------------------
extern "C" void kernel_launch(void* const* d_in, const int* in_sizes, int n_in,
                              void* d_out, int out_size, void* d_ws, size_t ws_size,
                              hipStream_t stream)
{
  const float* X    = (const float*)d_in[0];   // (8192, 512)
  const float* adj  = (const float*)d_in[1];   // (8192, 8192)
  const float* eye  = (const float*)d_in[2];   // (8192, 8192)
  const float* W    = (const float*)d_in[3];   // (64, 512)
  const float* bias = (const float*)d_in[4];   // (64,)
  const float* a1   = (const float*)d_in[5];   // (1, 64)
  const float* a2   = (const float*)d_in[6];   // (1, 64)
  float* out = (float*)d_out;                  // (8192, 64)

  // workspace layout (~11.3 MB total)
  char* ws = (char*)d_ws;
  const size_t MB = 1u << 20;
  float*  z     = (float*) (ws);                         // 2 MB   fp32 z
  __bf16* zbt   = (__bf16*)(ws + 2 * MB);                // 1 MB   bf16 z^T
  float*  Ep    = (float*) (ws + 3 * MB);                // 8 MB   adj@z partials
  float*  dsumP = (float*) (ws + 11 * MB);               // 128 KB degree partials
  float*  zi    = (float*) (ws + 11 * MB + 131072);      // 32 KB
  float*  zj    = (float*) (ws + 11 * MB + 163840);      // 32 KB
  float*  SallP = (float*) (ws + 11 * MB + 196608);      // 16 KB
  float*  Sall  = (float*) (ws + 11 * MB + 212992);      // 256 B

  (void)in_sizes; (void)n_in; (void)out_size; (void)ws_size;

  gemm1_z_kernel    <<<64,          256, 0, stream>>>(X, W, bias, z, zbt);
  row_zizj_kernel   <<<32,          256, 0, stream>>>(z, a1, a2, zi, zj);
  colsum_part_kernel<<<64,          256, 0, stream>>>(z, SallP);
  colsum_final_kernel<<<1,           64, 0, stream>>>(SallP, Sall);
  gemm2_attn_kernel <<<64 * KSPLIT, 256, 0, stream>>>(adj, zbt, Ep, dsumP);
  finalize_kernel   <<<N_ROWS,       64, 0, stream>>>(z, Ep, dsumP, zi, zj,
                                                      Sall, adj, eye, out);
}
